// CrossScaleAttention_30691836297752
// MI455X (gfx1250) — compile-verified, matrix operand order checked by reading, not run on hardware
//
#include <hip/hip_runtime.h>
#include <hip/hip_bf16.h>

typedef __attribute__((ext_vector_type(16))) _Float16 v16h;
typedef __attribute__((ext_vector_type(8)))  float    v8f;

#define NH 8
#define HD 32
#define HIDDEN 256

union V16U { uint4 q[2]; v16h v; };

// Operand from a [*,32]-f16 row (Q/K head-transposed layout).
// Lane's halves j=0..7 -> k = 8*hi + j ; j=8..15 -> k = 16 + 8*hi + (j-8).
__device__ __forceinline__ v16h load_row32(const _Float16* row, int hi) {
  V16U u;
  const uint4* p = (const uint4*)row;
  u.q[0] = p[hi];
  u.q[1] = p[2 + hi];
  return u.v;
}

// Operand whose K dim runs contiguously along a f16 row (P rows in LDS, V d-major rows):
// key blocks [kc+8*hi, +8) and [kc+16+8*hi, +8).
__device__ __forceinline__ v16h load_krow(const _Float16* row, int kc, int hi) {
  V16U u;
  u.q[0] = *(const uint4*)(row + kc + 8 * hi);
  u.q[1] = *(const uint4*)(row + kc + 16 + 8 * hi);
  return u.v;
}

// y[b,o,n] = bias[o] + sum_c w[o,c]*x[b,c,n]
// mode 0: f32 out [B,O,N]; mode 1: f16 out [B,O/32,N,32] (head-transposed); mode 2: f16 out [B,O,N]
__global__ __launch_bounds__(256) void proj_kernel(
    const float* __restrict__ x, const float* __restrict__ w, const float* __restrict__ bias,
    void* __restrict__ out, int C, int N, int mode)
{
  int n = blockIdx.x * 256 + threadIdx.x;
  int o = blockIdx.y;
  int b = blockIdx.z;
  int O = gridDim.y;
  const float* xr = x + (size_t)b * C * N + n;
  const float* wr = w + (size_t)o * C;
  float acc = bias[o];
#pragma unroll 4
  for (int c = 0; c < C; ++c) acc = fmaf(wr[c], xr[(size_t)c * N], acc);
  if (mode == 0) {
    ((float*)out)[((size_t)b * O + o) * N + n] = acc;
  } else if (mode == 1) {
    ((_Float16*)out)[(((size_t)b * NH + (o >> 5)) * N + n) * HD + (o & 31)] = (_Float16)acc;
  } else {
    ((_Float16*)out)[((size_t)b * O + o) * N + n] = (_Float16)acc;
  }
}

// Bilinear resize, half-pixel centers, edge clamp (align_corners=False, no antialias).
__global__ void resize_kernel(const float* __restrict__ in, float* __restrict__ out,
                              int B, int C, int Hi, int Wi, int Ho, int Wo)
{
  int idx = blockIdx.x * 256 + threadIdx.x;
  int total = B * C * Ho * Wo;
  if (idx >= total) return;
  int x = idx % Wo;
  int y = (idx / Wo) % Ho;
  int bc = idx / (Wo * Ho);
  float fy = (y + 0.5f) * (float)Hi / (float)Ho - 0.5f;
  float fx = (x + 0.5f) * (float)Wi / (float)Wo - 0.5f;
  int y0 = (int)floorf(fy); float dy = fy - (float)y0;
  int x0 = (int)floorf(fx); float dx = fx - (float)x0;
  int y1 = y0 + 1, x1 = x0 + 1;
  y0 = max(0, min(y0, Hi - 1)); y1 = max(0, min(y1, Hi - 1));
  x0 = max(0, min(x0, Wi - 1)); x1 = max(0, min(x1, Wi - 1));
  const float* p = in + (size_t)bc * Hi * Wi;
  float v00 = p[y0 * Wi + x0], v01 = p[y0 * Wi + x1];
  float v10 = p[y1 * Wi + x0], v11 = p[y1 * Wi + x1];
  out[idx] = (1.f - dy) * ((1.f - dx) * v00 + dx * v01) + dy * ((1.f - dx) * v10 + dx * v11);
}

// f32 [B,256,N] -> f16 [B,8,N,32] (head-transpose)
__global__ void cvt_tr_kernel(const float* __restrict__ in, _Float16* __restrict__ out, int N)
{
  int idx = blockIdx.x * 256 + threadIdx.x;
  int n = idx % N;
  int ch = (idx / N) % HIDDEN;
  int b = idx / (N * HIDDEN);
  out[(((size_t)b * NH + (ch >> 5)) * N + n) * HD + (ch & 31)] = (_Float16)in[idx];
}

__global__ void cvt_kernel(const float* __restrict__ in, _Float16* __restrict__ out, int total)
{
  int idx = blockIdx.x * 256 + threadIdx.x;
  if (idx < total) out[idx] = (_Float16)in[idx];
}

// One workgroup = one (b, head, 16-query tile). 8 waves; full key axis in LDS (f16).
__global__ __launch_bounds__(256) void attn_kernel(
    const _Float16* __restrict__ Q16,   // [B, NH, N, 32]
    const _Float16* __restrict__ K16,   // [B, NH, N, 32]
    const _Float16* __restrict__ V16T,  // [B, HIDDEN, N] (d-major)
    float* __restrict__ out,            // [B, HIDDEN, N]
    int N, float scale)
{
  extern __shared__ char smem[];
  _Float16* S   = (_Float16*)smem;                        // [16][N] scores -> probs
  float* Opart  = (float*)(smem + (size_t)16 * N * 2);    // [8][16*32] per-wave partials
  float* red    = Opart + 8 * 512;                        // [256]
  float* rowmax = red + 256;                              // [16]
  float* rowsum = rowmax + 16;                            // [16]

  const int tid  = threadIdx.x;
  const int wave = tid >> 5;
  const int lane = tid & 31;
  const int hi   = (lane >> 4) & 1;
  const int ln   = lane & 15;

  const int b = blockIdx.z, h = blockIdx.y;
  const int qbase = blockIdx.x << 4;
  const int bh = b * NH + h;

  // ---- scores: S[q, k] = scale * Q Kt ; one wmma per 16x16 tile (K dim = head_dim = 32) ----
  v16h aq = load_row32(Q16 + ((size_t)bh * N + qbase + ln) * HD, hi);
  const int span = N >> 3;        // keys per wave
  const int kt   = span >> 4;     // 16-wide key tiles per wave
  for (int t = 0; t < kt; ++t) {
    int kbase = wave * span + t * 16;
    v16h bk = load_row32(K16 + ((size_t)bh * N + kbase + ln) * HD, hi);
    v8f c = {};
    c = __builtin_amdgcn_wmma_f32_16x16x32_f16(false, aq, false, bk, (short)0, c, false, false);
#pragma unroll
    for (int r = 0; r < 8; ++r) {
      int m = r + 8 * hi;
      S[(size_t)m * N + kbase + ln] = (_Float16)(c[r] * scale);
    }
  }
  __syncthreads();

  // ---- softmax (f32 math over f16 scores), probs stored unnormalized; row sums kept ----
  {
    int row = tid >> 4, col = tid & 15;
    _Float16* sr = S + (size_t)row * N;
    float mx = -3.0e38f;
    for (int i = col; i < N; i += 16) mx = fmaxf(mx, (float)sr[i]);
    red[tid] = mx;
    __syncthreads();
    if (col == 0) {
      float m2 = red[row * 16];
      for (int j = 1; j < 16; ++j) m2 = fmaxf(m2, red[row * 16 + j]);
      rowmax[row] = m2;
    }
    __syncthreads();
    float rm = rowmax[row];
    float s = 0.f;
    for (int i = col; i < N; i += 16) {
      float e = __expf((float)sr[i] - rm);
      s += e;
      sr[i] = (_Float16)e;
    }
    red[tid] = s;
    __syncthreads();
    if (col == 0) {
      float s2 = 0.f;
      for (int j = 0; j < 16; ++j) s2 += red[row * 16 + j];
      rowsum[row] = s2;
    }
    __syncthreads();
  }

  // ---- O = P V : 2 wmma per 32-key chunk (d tiles 0..15, 16..31), per-wave accumulation ----
  v8f c0 = {}, c1 = {};
  const int chunks = span >> 5;
  const _Float16* prow = S + (size_t)ln * N;                                  // A: row = query ln
  const _Float16* v0 = V16T + ((size_t)b * HIDDEN + h * HD + ln) * N;         // B tile0: col d = ln
  const _Float16* v1 = V16T + ((size_t)b * HIDDEN + h * HD + 16 + ln) * N;    // B tile1: col d = 16+ln
  for (int u = 0; u < chunks; ++u) {
    int kc = wave * span + u * 32;
    v16h ap = load_krow(prow, kc, hi);
    v16h b0 = load_krow(v0, kc, hi);
    c0 = __builtin_amdgcn_wmma_f32_16x16x32_f16(false, ap, false, b0, (short)0, c0, false, false);
    v16h b1 = load_krow(v1, kc, hi);
    c1 = __builtin_amdgcn_wmma_f32_16x16x32_f16(false, ap, false, b1, (short)0, c1, false, false);
  }
  float* op = Opart + wave * 512;
#pragma unroll
  for (int r = 0; r < 8; ++r) {
    int m = r + 8 * hi;
    op[m * 32 + ln]      = c0[r];
    op[m * 32 + 16 + ln] = c1[r];
  }
  __syncthreads();

  // ---- cross-wave reduce, normalize, write [B,256,N] map ----
  for (int e = tid; e < 512; e += 256) {
    int m = e >> 5, d = e & 31;
    float s = 0.f;
#pragma unroll
    for (int w2 = 0; w2 < 8; ++w2) s += Opart[w2 * 512 + e];
    out[((size_t)b * HIDDEN + h * HD + d) * N + qbase + m] = s / rowsum[m];
  }
}

extern "C" void kernel_launch(void* const* d_in, const int* in_sizes, int n_in,
                              void* d_out, int out_size, void* d_ws, size_t ws_size,
                              hipStream_t stream)
{
  (void)in_sizes; (void)n_in; (void)out_size; (void)ws_size;
  const float* high = (const float*)d_in[0];
  const float* low  = (const float*)d_in[1];
  const float* q_hw = (const float*)d_in[2];  const float* q_hb = (const float*)d_in[3];
  const float* k_hw = (const float*)d_in[4];  const float* k_hb = (const float*)d_in[5];
  const float* v_hw = (const float*)d_in[6];  const float* v_hb = (const float*)d_in[7];
  const float* q_lw = (const float*)d_in[8];  const float* q_lb = (const float*)d_in[9];
  const float* k_lw = (const float*)d_in[10]; const float* k_lb = (const float*)d_in[11];
  const float* v_lw = (const float*)d_in[12]; const float* v_lb = (const float*)d_in[13];
  const float* o_hw = (const float*)d_in[14]; const float* o_hb = (const float*)d_in[15];
  const float* o_lw = (const float*)d_in[16]; const float* o_lb = (const float*)d_in[17];

  const int B = 2, Hh = 64, Wh = 64, Hl = 32, Wl = 32;
  const int Nh = 4096, Nl = 1024, Ch = 128, Cl = 256;
  const float scale = 0.17677669529663687f; // 32^-0.5

  char* ws = (char*)d_ws;
  size_t off = 0;
  auto alloc = [&](size_t bytes) -> char* {
    char* p = ws + off;
    off += (bytes + 255) & ~(size_t)255;
    return p;
  };
  float*    low_up = (float*)   alloc((size_t)B * Cl * Nh * 4);      // upsampled low feat
  _Float16* q_h16  = (_Float16*)alloc((size_t)B * NH * Nh * HD * 2); // [B,8,N,32]
  _Float16* k_l16  = (_Float16*)alloc((size_t)B * NH * Nh * HD * 2);
  _Float16* v_l16  = (_Float16*)alloc((size_t)B * HIDDEN * Nh * 2);  // d-major
  _Float16* q_l16  = (_Float16*)alloc((size_t)B * NH * Nl * HD * 2);
  float*    k_hf   = (float*)   alloc((size_t)B * HIDDEN * Nh * 4);  // k_high at 64x64
  float*    v_hf   = (float*)   alloc((size_t)B * HIDDEN * Nh * 4);
  float*    k_hr   = (float*)   alloc((size_t)B * HIDDEN * Nl * 4);  // resized to 32x32
  float*    v_hr   = (float*)   alloc((size_t)B * HIDDEN * Nl * 4);
  _Float16* k_h16  = (_Float16*)alloc((size_t)B * NH * Nl * HD * 2);
  _Float16* v_h16  = (_Float16*)alloc((size_t)B * HIDDEN * Nl * 2);
  float*    at_h   = (float*)   alloc((size_t)B * HIDDEN * Nh * 4);  // attention out maps
  float*    at_l   = (float*)   alloc((size_t)B * HIDDEN * Nl * 4);

  size_t smem_hi = (size_t)16 * Nh * 2 + 8 * 512 * 4 + (256 + 32) * 4;
  size_t smem_lo = (size_t)16 * Nl * 2 + 8 * 512 * 4 + (256 + 32) * 4;
  hipFuncSetAttribute(reinterpret_cast<const void*>(attn_kernel),
                      hipFuncAttributeMaxDynamicSharedMemorySize, (int)smem_hi);

  // ---- high branch: q from high, k/v from upsampled low ----
  {
    int total = B * Cl * Nh;
    resize_kernel<<<(total + 255) / 256, 256, 0, stream>>>(low, low_up, B, Cl, Hl, Wl, Hh, Wh);
  }
  proj_kernel<<<dim3(Nh / 256, HIDDEN, B), 256, 0, stream>>>(high,   q_hw, q_hb, q_h16, Ch, Nh, 1);
  proj_kernel<<<dim3(Nh / 256, HIDDEN, B), 256, 0, stream>>>(low_up, k_lw, k_lb, k_l16, Cl, Nh, 1);
  proj_kernel<<<dim3(Nh / 256, HIDDEN, B), 256, 0, stream>>>(low_up, v_lw, v_lb, v_l16, Cl, Nh, 2);
  attn_kernel<<<dim3(Nh / 16, NH, B), 256, smem_hi, stream>>>(q_h16, k_l16, v_l16, at_h, Nh, scale);
  proj_kernel<<<dim3(Nh / 256, Ch, B), 256, 0, stream>>>(at_h, o_hw, o_hb, (float*)d_out, HIDDEN, Nh, 0);

  // ---- low branch: q from low, k/v from downsampled high projections ----
  proj_kernel<<<dim3(Nl / 256, HIDDEN, B), 256, 0, stream>>>(low,  q_lw, q_lb, q_l16, Cl, Nl, 1);
  proj_kernel<<<dim3(Nh / 256, HIDDEN, B), 256, 0, stream>>>(high, k_hw, k_hb, k_hf, Ch, Nh, 0);
  proj_kernel<<<dim3(Nh / 256, HIDDEN, B), 256, 0, stream>>>(high, v_hw, v_hb, v_hf, Ch, Nh, 0);
  {
    int total = B * HIDDEN * Nl;
    resize_kernel<<<(total + 255) / 256, 256, 0, stream>>>(k_hf, k_hr, B, HIDDEN, Hh, Wh, Hl, Wl);
    resize_kernel<<<(total + 255) / 256, 256, 0, stream>>>(v_hf, v_hr, B, HIDDEN, Hh, Wh, Hl, Wl);
    cvt_tr_kernel<<<total / 256, 256, 0, stream>>>(k_hr, k_h16, Nl);
    cvt_kernel<<<total / 256, 256, 0, stream>>>(v_hr, v_h16, total);
  }
  attn_kernel<<<dim3(Nl / 16, NH, B), 256, smem_lo, stream>>>(q_l16, k_h16, v_h16, at_l, Nl, scale);
  proj_kernel<<<dim3(Nl / 256, HIDDEN, B), 256, 0, stream>>>(
      at_l, o_lw, o_lb, (float*)d_out + (size_t)B * Ch * Nh, HIDDEN, Nl, 0);
}
